// RQVAE_9629316677979
// MI455X (gfx1250) — matrix operand order repaired and checked by reading, not compiled, not run on hardware
//
#include <hip/hip_runtime.h>
#include <stdint.h>

// ---------------------------------------------------------------------------
// Types for CDNA5 WMMA / TDM
// ---------------------------------------------------------------------------
typedef __attribute__((ext_vector_type(16))) __bf16 v16bf;
typedef __attribute__((ext_vector_type(8)))  float  v8f;
typedef __attribute__((ext_vector_type(4)))  unsigned int v4u;
typedef __attribute__((ext_vector_type(8)))  int v8i;
typedef __attribute__((ext_vector_type(4)))  int v4i;

union Frag16 {
  v16bf    v;
  uint32_t w[8];
  uint4    q[2];
};
union G0u { v4u v; uint32_t w[4]; };
union G1u { v8i v; int32_t  w[8]; };

__device__ __forceinline__ uint32_t pk_bf16(float a, float b) {
  uint32_t ua = __float_as_uint(a), ub = __float_as_uint(b);
  ua += 0x7FFFu + ((ua >> 16) & 1u);   // round-to-nearest-even
  ub += 0x7FFFu + ((ub >> 16) & 1u);
  return (ua >> 16) | (ub & 0xFFFF0000u);
}
__device__ __forceinline__ uint16_t f2bf(float a) {
  uint32_t u = __float_as_uint(a);
  u += 0x7FFFu + ((u >> 16) & 1u);
  return (uint16_t)(u >> 16);
}

// ---------------------------------------------------------------------------
// TDM: DMA a 2D bf16 tile (rows x row_elems, global row stride stride_elems)
// into LDS at lds_off, inserting LDS padding after each row.
// Descriptor layout per CDNA5 ISA.
//   pad_interval encoding: 2^(v+1) dwords between pads
//   pad_amount  encoding: (v+1) dwords of pad
// Must be reached by one wave only (wave-uniform branch); tracked by TENSORcnt.
// This toolchain exposes the 6-arg builtin (g0, g1, g2, g3, g4, cpol);
// trailing zero-filled groups encode an (up to 5D) tensor with unused dims.
// ---------------------------------------------------------------------------
__device__ __forceinline__ void tdm_load_2d_bf16(
    uint32_t lds_off, const void* gptr, uint32_t row_elems, uint32_t rows,
    uint32_t stride_elems, uint32_t pad_interval_enc, uint32_t pad_amount_enc)
{
  const uint64_t ga = (uint64_t)(uintptr_t)gptr;
  G0u g0;
  g0.w[0] = 1u;                                   // count=1, user descriptor
  g0.w[1] = lds_off;                              // lds_addr (bytes)
  g0.w[2] = (uint32_t)ga;                         // global_addr[31:0]
  g0.w[3] = (uint32_t)(ga >> 32) | (2u << 30);    // global_addr[56:32] | type=2

  uint32_t w0 = (1u << 16);                       // data_size=1 -> 2 bytes
  w0 |= (1u << 20) | (pad_interval_enc << 22) | (pad_amount_enc << 25);
  const uint32_t td0 = stride_elems;              // tensor_dim0 (OOB guard)
  const uint32_t td1 = 0x3FFFFFFFu;               // tensor_dim1: effectively no OOB
  G1u g1;
  g1.w[0] = (int32_t)w0;
  g1.w[1] = (int32_t)((td0 & 0xFFFFu) << 16);                 // tensor_dim0[15:0]
  g1.w[2] = (int32_t)((td0 >> 16) | ((td1 & 0xFFFFu) << 16)); // dim0 hi | dim1 lo
  g1.w[3] = (int32_t)((td1 >> 16) | (row_elems << 16));       // dim1 hi | tile_dim0
  g1.w[4] = (int32_t)(rows & 0xFFFFu);                        // tile_dim1 (tile_dim2=0)
  g1.w[5] = (int32_t)stride_elems;                            // tensor_dim0_stride lo
  g1.w[6] = 0;
  g1.w[7] = 0;
  const v4i z4 = {0, 0, 0, 0};                    // groups 2/3: 2D tile
  const v8i z8 = {0, 0, 0, 0, 0, 0, 0, 0};        // trailing group: unused dims
  __builtin_amdgcn_tensor_load_to_lds(g0.v, g1.v, z4, z4, z8, 0);
}

__device__ __forceinline__ uint32_t lds_off_of(const void* p) {
  return (uint32_t)(uintptr_t)p;   // flat shared addr low 32 bits == LDS offset
}

// ---------------------------------------------------------------------------
// Prep kernels: weights -> transposed bf16; codebooks -> bf16 + ||c||^2
// ---------------------------------------------------------------------------
__global__ __launch_bounds__(256)
void cvt_transpose_bf16(const float* __restrict__ W, uint16_t* __restrict__ WT,
                        int K, int N) {
  const int idx = blockIdx.x * 256 + threadIdx.x;
  if (idx < K * N) {
    const int n = idx / K, k = idx % K;
    WT[idx] = f2bf(W[(size_t)k * N + n]);   // WT[n][k] = W[k][n]
  }
}

__global__ __launch_bounds__(256)
void prep_codebooks(const float* __restrict__ cbs, uint16_t* __restrict__ cbb,
                    float* __restrict__ cnb) {
  const int c = blockIdx.x * 256 + threadIdx.x;   // 0..1023 = L*K codes
  const float* src = cbs + (size_t)c * 64;
  float s = 0.0f;
#pragma unroll
  for (int d = 0; d < 64; ++d) {
    const float v = src[d];
    s += v * v;
    cbb[(size_t)c * 64 + d] = f2bf(v);
  }
  cnb[c] = s;
}

// ---------------------------------------------------------------------------
// bf16-WMMA GEMM:  C[M,N] = act(A[M,K] @ BT[N,K]^T + bias)
// BM=128, KT=32; 256 threads = 8 waves (4x2); wave owns 2 x (BN/32) 16x16 tiles.
// BT is pre-transposed bf16 so B tiles are row-major contiguous -> TDM DMA.
// A_F32:  A is f32, register-staged + packed to bf16 (first layer only).
// !A_F32: A is bf16, double-buffered TDM DMA overlapped with WMMA.
// LDS tile pitch = 20 dwords (32 bf16 data + 16B pad) from TDM pad descriptors.
// ---------------------------------------------------------------------------
template <int BN, bool RELU, bool A_F32, bool OUT_F32>
__global__ __launch_bounds__(256)
void gemm_bf16_wmma(const void* __restrict__ Av, const uint16_t* __restrict__ BT,
                    const float* __restrict__ bias, void* __restrict__ Cv,
                    int M, int Ncols, int K)
{
  constexpr int BM   = 128;
  constexpr int APD  = 20;        // dword pitch: 16 data + 4 pad
  constexpr int NT   = BN / 32;
  constexpr int NBUF = A_F32 ? 1 : 2;

  __shared__ uint32_t Asb[NBUF * BM * APD];
  __shared__ uint32_t Bsb[NBUF * BN * APD];

  const int tid  = threadIdx.x;
  const int lane = tid & 31;
  const int wid  = tid >> 5;
  const int wr   = wid & 3;
  const int wc   = wid >> 2;
  const int lm   = lane & 15;
  const int lh   = lane >> 4;
  const int bm   = blockIdx.y * BM;
  const int bn   = blockIdx.x * BN;

  v8f acc[2][NT];
#pragma unroll
  for (int mt = 0; mt < 2; ++mt)
#pragma unroll
    for (int nt = 0; nt < NT; ++nt) {
      v8f z = {};
      acc[mt][nt] = z;
    }

  const uint32_t aLds = lds_off_of(&Asb[0]);
  const uint32_t bLds = lds_off_of(&Bsb[0]);

  if (A_F32) {
    // ---------------- f32-A path (single buffer, B via TDM) ----------------
    const float* A = (const float*)Av;
    const int arow = tid >> 3, ac4 = tid & 7;
    for (int k0 = 0; k0 < K; k0 += 32) {
      float4 ar[4];
#pragma unroll
      for (int p = 0; p < 4; ++p)
        ar[p] = *(const float4*)(A + (size_t)(bm + arow + p * 32) * K + k0 + ac4 * 4);

      __syncthreads();   // previous iteration's LDS reads complete
      if (wid == 0)
        tdm_load_2d_bf16(bLds, BT + (size_t)bn * K + k0, 32, BN, K, 3, 3);
#pragma unroll
      for (int p = 0; p < 4; ++p) {
        uint32_t* d = &Asb[(arow + p * 32) * APD + ac4 * 2];
        d[0] = pk_bf16(ar[p].x, ar[p].y);
        d[1] = pk_bf16(ar[p].z, ar[p].w);
      }
      if (wid == 0) __builtin_amdgcn_s_wait_tensorcnt(0);
      __syncthreads();

      Frag16 af[2];
#pragma unroll
      for (int mt = 0; mt < 2; ++mt) {
        const int row = wr * 32 + mt * 16 + lm;
        af[mt].q[0] = *(const uint4*)&Asb[row * APD + lh * 4];
        af[mt].q[1] = *(const uint4*)&Asb[row * APD + lh * 4 + 8];
      }
      Frag16 bfr[NT];
#pragma unroll
      for (int nt = 0; nt < NT; ++nt) {
        const int col = wc * (BN / 2) + nt * 16 + lm;
        bfr[nt].q[0] = *(const uint4*)&Bsb[col * APD + lh * 4];
        bfr[nt].q[1] = *(const uint4*)&Bsb[col * APD + lh * 4 + 8];
      }
#pragma unroll
      for (int mt = 0; mt < 2; ++mt)
#pragma unroll
        for (int nt = 0; nt < NT; ++nt)
          acc[mt][nt] = __builtin_amdgcn_wmma_f32_16x16x32_bf16(
              false, af[mt].v, false, bfr[nt].v, (short)0, acc[mt][nt], false, false);
    }
  } else {
    // ------------- bf16-A path: double-buffered TDM for A and B -------------
    const uint16_t* A = (const uint16_t*)Av;
    const int iters = K / 32;
    if (wid == 0) {
      tdm_load_2d_bf16(aLds, A + (size_t)bm * K, 32, BM, K, 3, 3);
      tdm_load_2d_bf16(bLds, BT + (size_t)bn * K, 32, BN, K, 3, 3);
      __builtin_amdgcn_s_wait_tensorcnt(0);
    }
    __syncthreads();

    for (int i = 0; i < iters; ++i) {
      const int cur = i & 1;
      if (wid == 0 && i + 1 < iters) {   // prefetch next tiles while computing
        const int kn = (i + 1) * 32;
        tdm_load_2d_bf16(aLds + (1 - cur) * BM * APD * 4, A + (size_t)bm * K + kn,
                         32, BM, K, 3, 3);
        tdm_load_2d_bf16(bLds + (1 - cur) * BN * APD * 4, BT + (size_t)bn * K + kn,
                         32, BN, K, 3, 3);
      }

      const uint32_t* Ab = &Asb[cur * BM * APD];
      const uint32_t* Bb = &Bsb[cur * BN * APD];
      Frag16 af[2];
#pragma unroll
      for (int mt = 0; mt < 2; ++mt) {
        const int row = wr * 32 + mt * 16 + lm;
        af[mt].q[0] = *(const uint4*)&Ab[row * APD + lh * 4];
        af[mt].q[1] = *(const uint4*)&Ab[row * APD + lh * 4 + 8];
      }
      Frag16 bfr[NT];
#pragma unroll
      for (int nt = 0; nt < NT; ++nt) {
        const int col = wc * (BN / 2) + nt * 16 + lm;
        bfr[nt].q[0] = *(const uint4*)&Bb[col * APD + lh * 4];
        bfr[nt].q[1] = *(const uint4*)&Bb[col * APD + lh * 4 + 8];
      }
#pragma unroll
      for (int mt = 0; mt < 2; ++mt)
#pragma unroll
        for (int nt = 0; nt < NT; ++nt)
          acc[mt][nt] = __builtin_amdgcn_wmma_f32_16x16x32_bf16(
              false, af[mt].v, false, bfr[nt].v, (short)0, acc[mt][nt], false, false);

      __syncthreads();                                   // everyone done with cur
      if (wid == 0 && i + 1 < iters) __builtin_amdgcn_s_wait_tensorcnt(0);
      __syncthreads();                                   // next buffer DMA visible
    }
  }

  // ---- epilogue: bias + optional ReLU; f32 or bf16 output ----
#pragma unroll
  for (int nt = 0; nt < NT; ++nt) {
    const int col = bn + wc * (BN / 2) + nt * 16 + lm;
    const float bv = bias[col];
#pragma unroll
    for (int mt = 0; mt < 2; ++mt) {
      const int row0 = bm + wr * 32 + mt * 16 + lh * 8;
#pragma unroll
      for (int v = 0; v < 8; ++v) {
        float x = acc[mt][nt][v] + bv;
        if (RELU) x = fmaxf(x, 0.0f);
        if (OUT_F32) ((float*)Cv)[(size_t)(row0 + v) * Ncols + col] = x;
        else        ((uint16_t*)Cv)[(size_t)(row0 + v) * Ncols + col] = f2bf(x);
      }
    }
  }
}

// ---------------------------------------------------------------------------
// Fused residual quantizer. 256 threads = 8 waves, 128 rows (E=64).
// Residual in LDS (f32). Per level: codebook (pre-bf16) DMA'd by TDM with pad
// -> 72-bf16 pitch; wave scores 16 rows x 256 codes via WMMA (argmin needs
// only ||c||^2 - 2 r.c); cross-lane argmin w/ index tie-break; fused gather /
// residual update / x_q(bf16) / loss.
// ---------------------------------------------------------------------------
__global__ __launch_bounds__(256)
void rq_quantize(const float* __restrict__ zin, const float* __restrict__ cbs,
                 const uint16_t* __restrict__ cbb, const float* __restrict__ cnb,
                 uint16_t* __restrict__ xq, int* __restrict__ idxOut,
                 float* __restrict__ lossOut, int Nrows)
{
  constexpr int ROWS = 128, E = 64, KC = 256, LV = 4;
  constexpr int CBPD = 36;   // 32 data dwords + 4 pad dwords per code row

  __shared__ float    res[ROWS * E];
  __shared__ uint32_t cbl[KC * CBPD];
  __shared__ float    cn[KC];
  __shared__ int      idxs[ROWS];
  __shared__ float    redbuf[256];

  const int tid  = threadIdx.x;
  const int lane = tid & 31;
  const int wid  = tid >> 5;
  const int lm   = lane & 15;
  const int lh   = lane >> 4;
  const int base = blockIdx.x * ROWS;
  const uint32_t cbLds = lds_off_of(&cbl[0]);

  // load 128x64 f32 of z into LDS
#pragma unroll
  for (int p = 0; p < 8; ++p) {
    float4 v = ((const float4*)(zin + (size_t)base * E))[tid + p * 256];
    ((float4*)res)[tid + p * 256] = v;
  }

  float xqr[16][2];
#pragma unroll
  for (int r = 0; r < 16; ++r) { xqr[r][0] = 0.0f; xqr[r][1] = 0.0f; }
  float lossAcc = 0.0f;

  for (int l = 0; l < LV; ++l) {
    __syncthreads();  // res stable; previous level's cbl reads done

    // ---- stage codebook level l: TDM DMA (64 bf16 rows -> pitch 72 bf16) ----
    if (wid == 0)
      tdm_load_2d_bf16(cbLds, cbb + (size_t)l * KC * E, E, KC, E, 4, 3);
    cn[tid] = cnb[l * KC + tid];
    if (wid == 0) __builtin_amdgcn_s_wait_tensorcnt(0);
    __syncthreads();

    // ---- build A fragments (16 rows x 64 dims -> two K=32 frags) ----
    Frag16 af[2];
#pragma unroll
    for (int f = 0; f < 2; ++f) {
      const float* rp = &res[(wid * 16 + lm) * E + f * 32 + lh * 8];
#pragma unroll
      for (int j = 0; j < 4; ++j) {
        af[f].w[j]     = pk_bf16(rp[2 * j],      rp[2 * j + 1]);
        af[f].w[4 + j] = pk_bf16(rp[16 + 2 * j], rp[16 + 2 * j + 1]);
      }
    }

    float bestd[8];
    int   bestc[8];
#pragma unroll
    for (int v = 0; v < 8; ++v) { bestd[v] = 3.4e38f; bestc[v] = 0; }

    for (int t = 0; t < 16; ++t) {
      Frag16 bf0, bf1;
      const int cb0 = (t * 16 + lm) * CBPD;
      bf0.q[0] = *(const uint4*)&cbl[cb0 + lh * 4];
      bf0.q[1] = *(const uint4*)&cbl[cb0 + lh * 4 + 8];
      bf1.q[0] = *(const uint4*)&cbl[cb0 + 16 + lh * 4];
      bf1.q[1] = *(const uint4*)&cbl[cb0 + 16 + lh * 4 + 8];

      v8f s = {};
      s = __builtin_amdgcn_wmma_f32_16x16x32_bf16(false, af[0].v, false, bf0.v,
                                                  (short)0, s, false, false);
      s = __builtin_amdgcn_wmma_f32_16x16x32_bf16(false, af[1].v, false, bf1.v,
                                                  (short)0, s, false, false);
      const int   c   = t * 16 + lm;
      const float cnv = cn[c];
#pragma unroll
      for (int v = 0; v < 8; ++v) {
        const float d = cnv - 2.0f * s[v];
        if (d < bestd[v]) { bestd[v] = d; bestc[v] = c; }
      }
    }

    // ---- cross-lane argmin within each 16-lane half ----
#pragma unroll
    for (int v = 0; v < 8; ++v) {
      float bd = bestd[v];
      int   bc = bestc[v];
#pragma unroll
      for (int off = 8; off >= 1; off >>= 1) {
        const float od = __shfl_xor(bd, off, 32);
        const int   oc = __shfl_xor(bc, off, 32);
        if (od < bd || (od == bd && oc < bc)) { bd = od; bc = oc; }
      }
      if (lm == 0) {
        const int rloc = wid * 16 + lh * 8 + v;   // lanes 0 & 16
        idxs[rloc] = bc;
        idxOut[(size_t)(base + rloc) * LV + l] = bc;
      }
    }
    __syncthreads();

    // ---- gather / residual update / x_q / loss (f32 codebook for accuracy) --
    const float* cbg = cbs + (size_t)l * KC * E;
#pragma unroll
    for (int r = 0; r < 16; ++r) {
      const int rr = wid * 16 + r;
      const int c  = idxs[rr];
      const float e0 = cbg[(size_t)c * E + lane];
      const float e1 = cbg[(size_t)c * E + 32 + lane];
      const float r0 = res[rr * E + lane];
      const float r1 = res[rr * E + 32 + lane];
      lossAcc += (e0 - r0) * (e0 - r0) + (e1 - r1) * (e1 - r1);
      res[rr * E + lane]      = r0 - e0;
      res[rr * E + 32 + lane] = r1 - e1;
      xqr[r][0] += e0;
      xqr[r][1] += e1;
    }
  }

  // ---- write x_q as bf16 (decoder input uses the TDM path) ----
#pragma unroll
  for (int r = 0; r < 16; ++r) {
    const int rr = wid * 16 + r;
    xq[(size_t)(base + rr) * E + lane]      = f2bf(xqr[r][0]);
    xq[(size_t)(base + rr) * E + 32 + lane] = f2bf(xqr[r][1]);
  }

  // ---- block loss reduction + atomic ----
  redbuf[tid] = lossAcc;
  __syncthreads();
  for (int s = 128; s > 0; s >>= 1) {
    if (tid < s) redbuf[tid] += redbuf[tid + s];
    __syncthreads();
  }
  if (tid == 0) {
    const float scale = 1.25f / (4.0f * (float)Nrows * 64.0f);
    atomicAdd(lossOut, redbuf[0] * scale);
  }
}

// ---------------------------------------------------------------------------
// Host-side launcher
// ---------------------------------------------------------------------------
extern "C" void kernel_launch(void* const* d_in, const int* in_sizes, int n_in,
                              void* d_out, int out_size, void* d_ws, size_t ws_size,
                              hipStream_t stream) {
  (void)in_sizes; (void)n_in; (void)out_size; (void)ws_size;
  const int N = 131072;

  const float* ew0 = (const float*)d_in[0];
  const float* eb0 = (const float*)d_in[1];
  const float* ew1 = (const float*)d_in[2];
  const float* eb1 = (const float*)d_in[3];
  const float* ew2 = (const float*)d_in[4];
  const float* eb2 = (const float*)d_in[5];
  const float* dw0 = (const float*)d_in[6];
  const float* db0 = (const float*)d_in[7];
  const float* dw1 = (const float*)d_in[8];
  const float* db1 = (const float*)d_in[9];
  const float* dw2 = (const float*)d_in[10];
  const float* db2 = (const float*)d_in[11];
  const float* cbs = (const float*)d_in[12];
  const float* x   = (const float*)d_in[13];

  // workspace layout
  uint16_t* h1  = (uint16_t*)d_ws;            // N*512 bf16
  uint16_t* h2  = h1 + (size_t)N * 512;       // N*256 bf16
  float*    z   = (float*)(h2 + (size_t)N * 256);   // N*64 f32
  uint16_t* xq  = (uint16_t*)(z + (size_t)N * 64);  // N*64 bf16
  uint16_t* wt0 = xq + (size_t)N * 64;        // 512*768
  uint16_t* wt1 = wt0 + 512 * 768;            // 256*512
  uint16_t* wt2 = wt1 + 256 * 512;            // 64*256
  uint16_t* wt3 = wt2 + 64 * 256;             // 256*64
  uint16_t* wt4 = wt3 + 256 * 64;             // 512*256
  uint16_t* wt5 = wt4 + 512 * 256;            // 768*512
  uint16_t* cbb = wt5 + 768 * 512;            // 4*256*64
  float*    cnb = (float*)(cbb + 4 * 256 * 64); // 4*256
  uint16_t* g1  = h2;                         // reuse
  uint16_t* g2  = h1;                         // reuse

  float* out  = (float*)d_out;                // N*768
  float* loss = out + (size_t)N * 768;        // 1
  int*   idxO = (int*)(loss + 1);             // N*4 (int32 bits)

  const dim3 blk(256);
  const int  MB = N / 128;

  // ---- prep: transposed bf16 weights + bf16 codebooks & norms ----
  cvt_transpose_bf16<<<(768 * 512 + 255) / 256, blk, 0, stream>>>(ew0, wt0, 768, 512);
  cvt_transpose_bf16<<<(512 * 256 + 255) / 256, blk, 0, stream>>>(ew1, wt1, 512, 256);
  cvt_transpose_bf16<<<(256 *  64 + 255) / 256, blk, 0, stream>>>(ew2, wt2, 256, 64);
  cvt_transpose_bf16<<<( 64 * 256 + 255) / 256, blk, 0, stream>>>(dw0, wt3, 64, 256);
  cvt_transpose_bf16<<<(256 * 512 + 255) / 256, blk, 0, stream>>>(dw1, wt4, 256, 512);
  cvt_transpose_bf16<<<(512 * 768 + 255) / 256, blk, 0, stream>>>(dw2, wt5, 512, 768);
  prep_codebooks<<<4, blk, 0, stream>>>(cbs, cbb, cnb);

  // ---- encoder ----
  gemm_bf16_wmma<128, true,  true,  false><<<dim3(4, MB), blk, 0, stream>>>(x,  wt0, eb0, h1, N, 512, 768);
  gemm_bf16_wmma<128, true,  false, false><<<dim3(2, MB), blk, 0, stream>>>(h1, wt1, eb1, h2, N, 256, 512);
  gemm_bf16_wmma< 64, false, false, true ><<<dim3(1, MB), blk, 0, stream>>>(h2, wt2, eb2, z,  N,  64, 256);

  // ---- residual quantization ----
  (void)hipMemsetAsync(loss, 0, sizeof(float), stream);
  rq_quantize<<<MB, blk, 0, stream>>>(z, cbs, cbb, cnb, xq, idxO, loss, N);

  // ---- decoder ----
  gemm_bf16_wmma<128, true,  false, false><<<dim3(2, MB), blk, 0, stream>>>(xq, wt3, db0, g1, N, 256, 64);
  gemm_bf16_wmma<128, true,  false, false><<<dim3(4, MB), blk, 0, stream>>>(g1, wt4, db1, g2, N, 512, 256);
  gemm_bf16_wmma<128, false, false, true ><<<dim3(6, MB), blk, 0, stream>>>(g2, wt5, db2, out, N, 768, 512);
}
